// RSSM_75883482185805
// MI455X (gfx1250) — compile-verified, matrix operand order checked by reading, not run on hardware
//
#include <hip/hip_runtime.h>
#include <hip/hip_bf16.h>
#include <math.h>

// ---------------- problem constants ----------------
#define T_STEPS 50
#define BATCH   1024
#define DETER   600
#define STOC    32
#define EMB     1024
#define ACTD    6
#define MLP     600
#define NOUT    792          // DETER + 6*STOC

// padded GEMM dims (N multiple of 128, K multiple of 32)
#define KIN 64               // stoc+act = 38 -> 64
#define NIN 640              // 600 -> 640   (x width)
#define NG  1920             // 1800 -> 1920 (gi/gh width)
#define KD  608              // 600 -> 608   (deter/h width as K)
#define NH  640              // 600 -> 640   (p1/q1 width)
#define KH  640              // p1/q1 as K
#define KQ  1632             // 600+1024=1624 -> 1632
#define NPH 128              // head output 64 -> 128

typedef __attribute__((ext_vector_type(16))) __bf16 v16bf;
typedef __attribute__((ext_vector_type(8)))  __bf16 v8bf;
typedef __attribute__((ext_vector_type(8)))  float  v8f;

typedef __attribute__((address_space(3))) __bf16 lds_bf16;

__device__ __forceinline__ float sigf(float x)  { return 1.0f / (1.0f + __expf(-x)); }
__device__ __forceinline__ float eluf(float x)  { return x > 0.0f ? x : expm1f(x); }
__device__ __forceinline__ float softplusf(float x) {
  return x > 20.0f ? x : log1pf(__expf(x));
}

// ---------------------------------------------------------------------------
// WMMA GEMM:  Y[M,N] = epi( X[M,Kp] @ W[N,Kp]^T + bias[N] )
//   X bf16 (row-major, ldx), W bf16 (row-major, ldw), bias fp32
//   epi==0 : fp32 out      epi==1 : ELU -> bf16 out
// Block: 256 threads = 8 waves in 4(M) x 2(N); block tile 128x128,
// wave tile 32x64.  W tiles (128x32 bf16) staged to LDS with the CDNA5
// async global->LDS path (ASYNCcnt), double-buffered.
// grid = (N/128, M/128)
// ---------------------------------------------------------------------------
__global__ __launch_bounds__(256)
void gemm_wmma_bf16(const __bf16* __restrict__ X, int ldx,
                    const __bf16* __restrict__ W, int ldw,
                    const float* __restrict__ bias,
                    float* __restrict__ Yf, __bf16* __restrict__ Yb, int ldy,
                    int K, int epi) {
  __shared__ __bf16 Wt[2][128 * 32];     // 16 KB, double buffered

  const int tid  = threadIdx.x;
  const int wave = tid >> 5;
  const int lane = tid & 31;
  const int g    = lane >> 4;            // lane group (0/1)
  const int mi   = lane & 15;
  const int wm   = wave >> 1;            // 0..3
  const int wn   = wave & 1;             // 0..1

  const int n0 = blockIdx.x * 128;               // block col base
  const int mw = blockIdx.y * 128 + wm * 32;     // wave row base
  const int nw = n0 + wn * 64;                   // wave col base

  v8f acc[2][4];
  #pragma unroll
  for (int s = 0; s < 2; ++s)
    #pragma unroll
    for (int j = 0; j < 4; ++j) acc[s][j] = (v8f){0,0,0,0,0,0,0,0};

  // async-stage one 128x32 W tile: 512 x 16B chunks, 2 chunks per thread.
  // The LDS offset is derived from the real address of Wt via an
  // addrspace(3) ptrtoint, so Wt escapes into the asm and the compiler must
  // assume the asm writes it (otherwise the later ds_loads get folded to
  // undef, as observed in round 2).
  auto stage = [&](int buf, int k0) {
    #pragma unroll
    for (int c = 0; c < 2; ++c) {
      const int chunk = tid * 2 + c;
      const int r     = chunk >> 2;            // 0..127
      const int col   = (chunk & 3) * 8;       // 0,8,16,24
      const __bf16* src = W + (size_t)(n0 + r) * ldw + k0 + col;
      __bf16* dst_gen = &Wt[buf][chunk * 8];
      lds_bf16* dst3  = (lds_bf16*)dst_gen;
      const unsigned loff = (unsigned)(unsigned long long)dst3;
      const unsigned long long ga = (unsigned long long)src;
      asm volatile("global_load_async_to_lds_b128 %0, %1, off"
                   :: "v"(loff), "v"(ga), "v"(dst_gen) : "memory");
    }
  };

  stage(0, 0);
  int buf = 0;
  for (int k0 = 0; k0 < K; k0 += 32) {
    asm volatile("s_wait_asynccnt 0" ::: "memory");
    __syncthreads();                      // staged tile visible to all waves
    if (k0 + 32 < K) stage(buf ^ 1, k0 + 32);   // overlap next fill with WMMAs

    // two A fragments (rows mw..+16, mw+16..+32), CDNA5 16-bit A layout
    v16bf a[2];
    #pragma unroll
    for (int s = 0; s < 2; ++s) {
      const __bf16* arow = X + (size_t)(mw + s * 16 + mi) * ldx + k0 + g * 8;
      v8bf lo = *(const v8bf*)arow;
      v8bf hi = *(const v8bf*)(arow + 16);
      #pragma unroll
      for (int i = 0; i < 8; ++i) { a[s][i] = lo[i]; a[s][8 + i] = hi[i]; }
    }

    #pragma unroll
    for (int j = 0; j < 4; ++j) {
      const __bf16* brow = &Wt[buf][(wn * 64 + j * 16 + mi) * 32 + g * 8];
      v8bf lo = *(const v8bf*)brow;
      v8bf hi = *(const v8bf*)(brow + 16);
      v16bf b;
      #pragma unroll
      for (int i = 0; i < 8; ++i) { b[i] = lo[i]; b[8 + i] = hi[i]; }
      acc[0][j] = __builtin_amdgcn_wmma_f32_16x16x32_bf16(
                      false, a[0], false, b, (short)0, acc[0][j], false, false);
      acc[1][j] = __builtin_amdgcn_wmma_f32_16x16x32_bf16(
                      false, a[1], false, b, (short)0, acc[1][j], false, false);
    }
    buf ^= 1;
  }

  // epilogue: C/D layout -> lane holds N = mi, VGPR v holds M = g*8 + v
  #pragma unroll
  for (int s = 0; s < 2; ++s) {
    #pragma unroll
    for (int j = 0; j < 4; ++j) {
      const int n  = nw + j * 16 + mi;
      const float bv = bias[n];
      #pragma unroll
      for (int v = 0; v < 8; ++v) {
        const int m = mw + s * 16 + g * 8 + v;
        float r = acc[s][j][v] + bv;
        if (epi == 1) {
          Yb[(size_t)m * ldy + n] = (__bf16)eluf(r);
        } else {
          Yf[(size_t)m * ldy + n] = r;
        }
      }
    }
  }
}

// ---------------------------------------------------------------------------
// weight fp32 -> bf16 with zero padding:  dst[Np,Kp]  from src[N,K]
// ---------------------------------------------------------------------------
__global__ void k_pad_w(const float* __restrict__ src, __bf16* __restrict__ dst,
                        int N, int K, int Kp, int total) {
  int idx = blockIdx.x * blockDim.x + threadIdx.x;
  if (idx >= total) return;
  int n = idx / Kp, k = idx - n * Kp;
  dst[idx] = (n < N && k < K) ? (__bf16)src[n * K + k] : (__bf16)0.0f;
}

__global__ void k_pad_b(const float* __restrict__ src, float* __restrict__ dst,
                        int N, int Np) {
  int idx = blockIdx.x * blockDim.x + threadIdx.x;
  if (idx >= Np) return;
  dst[idx] = (idx < N) ? src[idx] : 0.0f;
}

// init deter (fp32 + padded bf16) and stoc from inputs
__global__ void k_init_state(const float* __restrict__ init_deter,
                             const float* __restrict__ init_stoc,
                             float* __restrict__ deterf, __bf16* __restrict__ deterb,
                             float* __restrict__ stocf) {
  int idx = blockIdx.x * blockDim.x + threadIdx.x;     // B*KD
  if (idx >= BATCH * KD) return;
  int b = idx / KD, o = idx - b * KD;
  if (o < DETER) {
    float v = init_deter[b * DETER + o];
    deterf[b * DETER + o] = v;
    deterb[(size_t)b * KD + o] = (__bf16)v;
  } else {
    deterb[(size_t)b * KD + o] = (__bf16)0.0f;
  }
  if (o < STOC) stocf[b * STOC + o] = init_stoc[b * STOC + o];
}

// build xin[b, KIN] = bf16(concat(stoc*nt, action)), zero padded
__global__ void k_build_xin(const float* __restrict__ stocf,
                            const float* __restrict__ actions,   // [T,B,ACTD]
                            const float* __restrict__ nonterm,   // [T,B,1]
                            __bf16* __restrict__ xin, int t) {
  int idx = blockIdx.x * blockDim.x + threadIdx.x;     // B*KIN
  if (idx >= BATCH * KIN) return;
  int b = idx / KIN, j = idx - b * KIN;
  float v = 0.0f;
  if (j < STOC) {
    v = stocf[b * STOC + j] * nonterm[(size_t)t * BATCH + b];
  } else if (j < STOC + ACTD) {
    v = actions[((size_t)t * BATCH + b) * ACTD + (j - STOC)];
  }
  xin[(size_t)b * KIN + j] = (__bf16)v;
}

// GRU gates: consumes gi/gh [B,NG] fp32, old deter; writes h everywhere needed
__global__ void k_gru_gates(const float* __restrict__ gi, const float* __restrict__ gh,
                            float* __restrict__ deterf, __bf16* __restrict__ deterb,
                            __bf16* __restrict__ hq, float* __restrict__ out, int t) {
  int idx = blockIdx.x * blockDim.x + threadIdx.x;     // B*KD
  if (idx >= BATCH * KD) return;
  int b = idx / KD, o = idx - b * KD;
  if (o >= DETER) { deterb[(size_t)b * KD + o] = (__bf16)0.0f; return; }
  const float* gib = gi + (size_t)b * NG;
  const float* ghb = gh + (size_t)b * NG;
  float r = sigf(gib[o]            + ghb[o]);
  float z = sigf(gib[DETER + o]    + ghb[DETER + o]);
  float n = tanhf(gib[2 * DETER + o] + r * ghb[2 * DETER + o]);
  float hp = deterf[b * DETER + o];
  float h  = (1.0f - z) * n + z * hp;
  deterf[b * DETER + o]       = h;
  deterb[(size_t)b * KD + o]  = (__bf16)h;
  hq[(size_t)b * KQ + o]      = (__bf16)h;
  out[((size_t)t * BATCH + b) * NOUT + o] = h;
}

// copy emb[t] into hq[:, DETER:DETER+EMB] (bf16), zero tail padding
__global__ void k_emb_hq(const float* __restrict__ emb, __bf16* __restrict__ hq, int t) {
  int idx = blockIdx.x * blockDim.x + threadIdx.x;     // B*(KQ-DETER)
  const int W = KQ - DETER;                            // 1032
  if (idx >= BATCH * W) return;
  int b = idx / W, j = idx - b * W;
  float v = (j < EMB) ? emb[((size_t)t * BATCH + b) * EMB + j] : 0.0f;
  hq[(size_t)b * KQ + DETER + j] = (__bf16)v;
}

// heads: pp/qq [B,NPH] fp32 -> mu/std/sample, update stoc
__global__ void k_heads(const float* __restrict__ pp, const float* __restrict__ qq,
                        const float* __restrict__ noise_p, const float* __restrict__ noise_q,
                        float* __restrict__ stocf, float* __restrict__ out, int t) {
  int idx = blockIdx.x * blockDim.x + threadIdx.x;     // B*STOC
  if (idx >= BATCH * STOC) return;
  int b = idx / STOC, j = idx - b * STOC;
  float pmu = pp[b * NPH + j];
  float pst = softplusf(pp[b * NPH + STOC + j]) + 0.1f;
  float psm = pmu + pst * noise_p[((size_t)t * BATCH + b) * STOC + j];
  float qmu = qq[b * NPH + j];
  float qst = softplusf(qq[b * NPH + STOC + j]) + 0.1f;
  float qsm = qmu + qst * noise_q[((size_t)t * BATCH + b) * STOC + j];
  float* o = out + ((size_t)t * BATCH + b) * NOUT;
  o[DETER + 0 * STOC + j] = pmu;
  o[DETER + 1 * STOC + j] = pst;
  o[DETER + 2 * STOC + j] = psm;
  o[DETER + 3 * STOC + j] = qmu;
  o[DETER + 4 * STOC + j] = qst;
  o[DETER + 5 * STOC + j] = qsm;
  stocf[b * STOC + j] = qsm;
}

// ---------------------------------------------------------------------------
static inline size_t al256(size_t x) { return (x + 255) & ~(size_t)255; }

extern "C" void kernel_launch(void* const* d_in, const int* in_sizes, int n_in,
                              void* d_out, int out_size, void* d_ws, size_t ws_size,
                              hipStream_t stream) {
  const float* actions  = (const float*)d_in[0];
  const float* nonterm  = (const float*)d_in[1];
  const float* emb      = (const float*)d_in[2];
  const float* noise_p  = (const float*)d_in[3];
  const float* noise_q  = (const float*)d_in[4];
  const float* initd    = (const float*)d_in[5];
  const float* inits    = (const float*)d_in[6];
  const float* W_in = (const float*)d_in[7];  const float* b_in = (const float*)d_in[8];
  const float* W_ih = (const float*)d_in[9];  const float* b_ih = (const float*)d_in[10];
  const float* W_hh = (const float*)d_in[11]; const float* b_hh = (const float*)d_in[12];
  const float* Wp1  = (const float*)d_in[13]; const float* bp1  = (const float*)d_in[14];
  const float* Wp2  = (const float*)d_in[15]; const float* bp2  = (const float*)d_in[16];
  const float* Wq1  = (const float*)d_in[17]; const float* bq1  = (const float*)d_in[18];
  const float* Wq2  = (const float*)d_in[19]; const float* bq2  = (const float*)d_in[20];
  float* out = (float*)d_out;

  // ---- workspace bump allocation ----
  char* p = (char*)d_ws;
  auto take = [&](size_t bytes) { char* r = p; p += al256(bytes); return (void*)r; };
  __bf16* wbf_in = (__bf16*)take((size_t)NIN * KIN * 2);
  __bf16* wbf_ih = (__bf16*)take((size_t)NG  * KD  * 2);
  __bf16* wbf_hh = (__bf16*)take((size_t)NG  * KD  * 2);
  __bf16* wbf_p1 = (__bf16*)take((size_t)NH  * KD  * 2);
  __bf16* wbf_p2 = (__bf16*)take((size_t)NPH * KH  * 2);
  __bf16* wbf_q1 = (__bf16*)take((size_t)NH  * KQ  * 2);
  __bf16* wbf_q2 = (__bf16*)take((size_t)NPH * KH  * 2);
  float*  bb_in  = (float*)take(NIN * 4);
  float*  bb_ih  = (float*)take(NG  * 4);
  float*  bb_hh  = (float*)take(NG  * 4);
  float*  bb_p1  = (float*)take(NH  * 4);
  float*  bb_q1  = (float*)take(NH  * 4);
  float*  bb_p2  = (float*)take(NPH * 4);
  float*  bb_q2  = (float*)take(NPH * 4);
  __bf16* xin    = (__bf16*)take((size_t)BATCH * KIN * 2);
  __bf16* xbuf   = (__bf16*)take((size_t)BATCH * NIN * 2);
  __bf16* deterb = (__bf16*)take((size_t)BATCH * KD  * 2);
  float*  deterf = (float*) take((size_t)BATCH * DETER * 4);
  float*  stocf  = (float*) take((size_t)BATCH * STOC * 4);
  float*  gibuf  = (float*) take((size_t)BATCH * NG * 4);
  float*  ghbuf  = (float*) take((size_t)BATCH * NG * 4);
  __bf16* hq     = (__bf16*)take((size_t)BATCH * KQ * 2);
  __bf16* p1buf  = (__bf16*)take((size_t)BATCH * NH * 2);
  __bf16* q1buf  = (__bf16*)take((size_t)BATCH * NH * 2);
  float*  ppbuf  = (float*) take((size_t)BATCH * NPH * 4);
  float*  qqbuf  = (float*) take((size_t)BATCH * NPH * 4);
  (void)ws_size; (void)n_in; (void)in_sizes; (void)out_size;

  const int TB = 256;
  auto g1 = [&](int total) { return dim3((total + TB - 1) / TB); };

  // ---- weight conversion (bf16, zero padded) ----
  k_pad_w<<<g1(NIN * KIN), TB, 0, stream>>>(W_in, wbf_in, DETER, STOC + ACTD, KIN, NIN * KIN);
  k_pad_w<<<g1(NG  * KD ), TB, 0, stream>>>(W_ih, wbf_ih, 3 * DETER, DETER, KD, NG * KD);
  k_pad_w<<<g1(NG  * KD ), TB, 0, stream>>>(W_hh, wbf_hh, 3 * DETER, DETER, KD, NG * KD);
  k_pad_w<<<g1(NH  * KD ), TB, 0, stream>>>(Wp1,  wbf_p1, MLP, DETER, KD, NH * KD);
  k_pad_w<<<g1(NPH * KH ), TB, 0, stream>>>(Wp2,  wbf_p2, 2 * STOC, MLP, KH, NPH * KH);
  k_pad_w<<<g1(NH  * KQ ), TB, 0, stream>>>(Wq1,  wbf_q1, MLP, DETER + EMB, KQ, NH * KQ);
  k_pad_w<<<g1(NPH * KH ), TB, 0, stream>>>(Wq2,  wbf_q2, 2 * STOC, MLP, KH, NPH * KH);
  k_pad_b<<<g1(NIN), TB, 0, stream>>>(b_in, bb_in, DETER, NIN);
  k_pad_b<<<g1(NG ), TB, 0, stream>>>(b_ih, bb_ih, 3 * DETER, NG);
  k_pad_b<<<g1(NG ), TB, 0, stream>>>(b_hh, bb_hh, 3 * DETER, NG);
  k_pad_b<<<g1(NH ), TB, 0, stream>>>(bp1,  bb_p1, MLP, NH);
  k_pad_b<<<g1(NH ), TB, 0, stream>>>(bq1,  bb_q1, MLP, NH);
  k_pad_b<<<g1(NPH), TB, 0, stream>>>(bp2,  bb_p2, 2 * STOC, NPH);
  k_pad_b<<<g1(NPH), TB, 0, stream>>>(bq2,  bb_q2, 2 * STOC, NPH);

  k_init_state<<<g1(BATCH * KD), TB, 0, stream>>>(initd, inits, deterf, deterb, stocf);

  // ---- sequential rollout ----
  for (int t = 0; t < T_STEPS; ++t) {
    k_build_xin<<<g1(BATCH * KIN), TB, 0, stream>>>(stocf, actions, nonterm, xin, t);

    // x = ELU(xin @ W_in^T + b_in)            [B, NIN] bf16
    gemm_wmma_bf16<<<dim3(NIN / 128, BATCH / 128), 256, 0, stream>>>(
        xin, KIN, wbf_in, KIN, bb_in, nullptr, xbuf, NIN, KIN, 1);

    // gi = x @ W_ih^T + b_ih                  [B, NG] fp32
    gemm_wmma_bf16<<<dim3(NG / 128, BATCH / 128), 256, 0, stream>>>(
        xbuf, NIN, wbf_ih, KD, bb_ih, gibuf, nullptr, NG, KD, 0);

    // gh = deter @ W_hh^T + b_hh              [B, NG] fp32
    gemm_wmma_bf16<<<dim3(NG / 128, BATCH / 128), 256, 0, stream>>>(
        deterb, KD, wbf_hh, KD, bb_hh, ghbuf, nullptr, NG, KD, 0);

    // gates -> h (deterf, deterb, hq[:, :600], out[t, :, :600])
    k_gru_gates<<<g1(BATCH * KD), TB, 0, stream>>>(gibuf, ghbuf, deterf, deterb, hq, out, t);
    k_emb_hq<<<g1(BATCH * (KQ - DETER)), TB, 0, stream>>>(emb, hq, t);

    // prior head
    gemm_wmma_bf16<<<dim3(NH / 128, BATCH / 128), 256, 0, stream>>>(
        deterb, KD, wbf_p1, KD, bb_p1, nullptr, p1buf, NH, KD, 1);
    gemm_wmma_bf16<<<dim3(NPH / 128, BATCH / 128), 256, 0, stream>>>(
        p1buf, NH, wbf_p2, KH, bb_p2, ppbuf, nullptr, NPH, KH, 0);

    // posterior head
    gemm_wmma_bf16<<<dim3(NH / 128, BATCH / 128), 256, 0, stream>>>(
        hq, KQ, wbf_q1, KQ, bb_q1, nullptr, q1buf, NH, KQ, 1);
    gemm_wmma_bf16<<<dim3(NPH / 128, BATCH / 128), 256, 0, stream>>>(
        q1buf, NH, wbf_q2, KH, bb_q2, qqbuf, nullptr, NPH, KH, 0);

    k_heads<<<g1(BATCH * STOC), TB, 0, stream>>>(ppbuf, qqbuf, noise_p, noise_q, stocf, out, t);
  }
}
